// backbone_model_87522843558355
// MI455X (gfx1250) — compile-verified, matrix operand order checked by reading, not run on hardware
//
#include <hip/hip_runtime.h>

typedef __attribute__((ext_vector_type(16))) __bf16 v16bf;
typedef __attribute__((ext_vector_type(8)))  __bf16 v8bf;
typedef __attribute__((ext_vector_type(8)))  float  v8f;

#define Bb 8
#define Tt 12
#define Nn 512
#define Dd 128
#define Hh 8
#define Hd 16
#define Ffm 256
#define Mm (Bb*Tt*Nn)   /* 49152 */
#define BTn (Bb*Tt)     /* 96 */

static __device__ __forceinline__ v16bf mk16(v8bf lo, v8bf hi) {
  v16bf r;
#pragma unroll
  for (int i = 0; i < 8; ++i) { r[i] = lo[i]; r[i + 8] = hi[i]; }
  return r;
}
static __device__ __forceinline__ v16bf mk16z(v8bf lo) {
  v16bf r;
#pragma unroll
  for (int i = 0; i < 8; ++i) { r[i] = lo[i]; r[i + 8] = (__bf16)0.0f; }
  return r;
}

// ---- DPP butterfly reductions within each 16-lane row (VALU only, no LDS) ----
// quad_perm(1,0,3,2)=0xB1, quad_perm(2,3,0,1)=0x4E, row_half_mirror=0x141, row_mirror=0x140
template <int CTRL>
static __device__ __forceinline__ float dpp_get(float v) {
  int s = __builtin_amdgcn_update_dpp(0, __float_as_int(v), CTRL, 0xF, 0xF, true);
  return __int_as_float(s);
}
static __device__ __forceinline__ float rowmax16(float v) {
  v = fmaxf(v, dpp_get<0xB1>(v));
  v = fmaxf(v, dpp_get<0x4E>(v));
  v = fmaxf(v, dpp_get<0x141>(v));
  v = fmaxf(v, dpp_get<0x140>(v));
  return v;
}
static __device__ __forceinline__ float rowadd16(float v) {
  v = v + dpp_get<0xB1>(v);
  v = v + dpp_get<0x4E>(v);
  v = v + dpp_get<0x141>(v);
  v = v + dpp_get<0x140>(v);
  return v;
}

// ---------------- conversion / transpose ----------------
__global__ void k_f32_to_bf16(const float* __restrict__ in, __bf16* __restrict__ out, int n) {
  int i = blockIdx.x * 256 + threadIdx.x;
  if (i < n) out[i] = (__bf16)in[i];
}

// in: [rows, cols] f32 row-major  ->  out: [cols, rows] bf16 row-major
__global__ void k_transpose_bf16(const float* __restrict__ in, __bf16* __restrict__ out,
                                 int rows, int cols) {
  int i = blockIdx.x * 256 + threadIdx.x;
  if (i < rows * cols) {
    int r = i / cols, c = i % cols;
    out[(size_t)c * rows + r] = (__bf16)in[i];
  }
}

// ---------------- GEMM: C[M,Nout] = A[M,K](bf16) x Bt[Nout,K]^T + bias ----------------
// One wave computes a 32x64 tile: 2 M-tiles x 4 N-tiles, B-operand reused across M-tiles.
#define EPI_BF     0
#define EPI_VT     1
#define EPI_F32RES 2
#define EPI_RELU   3

template <int EPI>
__global__ void gemm_kernel(const __bf16* __restrict__ A, const __bf16* __restrict__ Bt,
                            const float* __restrict__ bias, int K, int Nout, float scale,
                            __bf16* __restrict__ out_bf, const float* __restrict__ res,
                            float* __restrict__ out_f) {
  const int lane = threadIdx.x & 31;
  const int wv = threadIdx.x >> 5;
  const int ngroups = Nout >> 6;
  const int w = blockIdx.x * 8 + wv;
  const int m0 = (w / ngroups) * 32;
  const int n0 = (w % ngroups) * 64;
  const int lr = lane & 15;
  const int g = lane >> 4;

  v8f acc[2][4];
#pragma unroll
  for (int s = 0; s < 2; ++s)
#pragma unroll
    for (int t = 0; t < 4; ++t)
#pragma unroll
      for (int j = 0; j < 8; ++j) acc[s][t][j] = 0.0f;

  for (int kk = 0; kk < K; kk += 32) {
    v16bf av[2];
#pragma unroll
    for (int s = 0; s < 2; ++s) {
      const __bf16* arow = A + (size_t)(m0 + s * 16 + lr) * K + kk + 8 * g;
      av[s] = mk16(*(const v8bf*)arow, *(const v8bf*)(arow + 16));
    }
#pragma unroll
    for (int t = 0; t < 4; ++t) {
      const __bf16* brow = Bt + (size_t)(n0 + t * 16 + lr) * K + kk + 8 * g;
      v16bf bvv = mk16(*(const v8bf*)brow, *(const v8bf*)(brow + 16));
      acc[0][t] = __builtin_amdgcn_wmma_f32_16x16x32_bf16(false, av[0], false, bvv,
                                                          (short)0, acc[0][t], false, false);
      acc[1][t] = __builtin_amdgcn_wmma_f32_16x16x32_bf16(false, av[1], false, bvv,
                                                          (short)0, acc[1][t], false, false);
    }
  }

#pragma unroll
  for (int s = 0; s < 2; ++s) {
    const int ms = m0 + s * 16;
#pragma unroll
    for (int t = 0; t < 4; ++t) {
      const int col = n0 + t * 16 + lr;
      const float bia = bias[col];
      if (EPI == EPI_VT) {
        const int bt = ms / Nn;
        const int nbase = (ms % Nn) + 8 * g;
        const int h = col >> 4, hd = col & 15;
        v8bf pack;
#pragma unroll
        for (int j = 0; j < 8; ++j) pack[j] = (__bf16)(acc[s][t][j] + bia);
        *(v8bf*)(out_bf + ((size_t)(bt * Hh + h) * Hd + hd) * Nn + nbase) = pack;
      } else {
#pragma unroll
        for (int j = 0; j < 8; ++j) {
          const int row = ms + j + 8 * g;
          const size_t idx = (size_t)row * Nout + col;
          const float v = acc[s][t][j] + bia;
          if (EPI == EPI_BF)          out_bf[idx] = (__bf16)(v * scale);
          else if (EPI == EPI_RELU)   out_bf[idx] = (__bf16)fmaxf(v, 0.0f);
          else if (EPI == EPI_F32RES) out_f[idx] = v + res[idx];
        }
      }
    }
  }
}

// ---------------- attention: one wave per (bt, h, 16-query block), 64 keys/iter -------
// q pre-scaled by 1/sqrt(Hd). q,k row-major [bt*Nn+n][Dd]; vT = [bt][h][hd][n].
__global__ void attn_kernel(const __bf16* __restrict__ q, const __bf16* __restrict__ k,
                            const __bf16* __restrict__ vT, __bf16* __restrict__ out) {
  __shared__ __align__(16) __bf16 psm[8 * 16 * 64];  // 16KB: per-wave 16x64 P tile
  const int lane = threadIdx.x & 31;
  const int wv = threadIdx.x >> 5;
  const int w = blockIdx.x * 8 + wv;
  const int qb = w & 31;
  const int h = (w >> 5) & 7;
  const int bt = w >> 8;
  const int m0 = qb * 16;
  const int lr = lane & 15;
  const int g = lane >> 4;
  __bf16* pbase = psm + wv * (16 * 64);

  const __bf16* qrow = q + (size_t)(bt * Nn + m0 + lr) * Dd + h * Hd + 8 * g;
  v16bf qa = mk16z(*(const v8bf*)qrow);

  float Mr[8], l[8];
  v8f acc;
#pragma unroll
  for (int j = 0; j < 8; ++j) { Mr[j] = -1e30f; l[j] = 0.0f; acc[j] = 0.0f; }

  for (int kt0 = 0; kt0 < Nn; kt0 += 64) {
    // ---- scores for 4 key tiles (K = Hd = 16, zero-padded to 32) ----
    v8f S[4];
#pragma unroll
    for (int tt = 0; tt < 4; ++tt) {
      const __bf16* krow = k + (size_t)(bt * Nn + kt0 + tt * 16 + lr) * Dd + h * Hd + 8 * g;
      v16bf kb = mk16z(*(const v8bf*)krow);
      v8f zero;
#pragma unroll
      for (int j = 0; j < 8; ++j) zero[j] = 0.0f;
      S[tt] = __builtin_amdgcn_wmma_f32_16x16x32_bf16(false, qa, false, kb,
                                                      (short)0, zero, false, false);
    }
    // ---- online softmax over 64 keys (one DPP reduction pass) ----
#pragma unroll
    for (int j = 0; j < 8; ++j) {
      float mj = fmaxf(fmaxf(S[0][j], S[1][j]), fmaxf(S[2][j], S[3][j]));
      mj = rowmax16(mj);
      const float Mn = fmaxf(Mr[j], mj);
      const float alpha = __expf(Mr[j] - Mn);
      Mr[j] = Mn;
      float rs = 0.0f;
#pragma unroll
      for (int tt = 0; tt < 4; ++tt) {
        const float p = __expf(S[tt][j] - Mn);
        rs += p;
        pbase[(j + 8 * g) * 64 + tt * 16 + lr] = (__bf16)p;  // C-layout scatter
      }
      rs = rowadd16(rs);
      l[j] = l[j] * alpha + rs;
      acc[j] *= alpha;
    }
    asm volatile("s_wait_dscnt 0" ::: "memory");  // same-wave DS in-order; drain stores
    // ---- P x V : two full-K (32-key) WMMAs ----
#pragma unroll
    for (int c = 0; c < 2; ++c) {
      const __bf16* prow = pbase + lr * 64 + c * 32 + 8 * g;
      v16bf pa = mk16(*(const v8bf*)prow, *(const v8bf*)(prow + 16));
      const __bf16* vrow = vT + ((size_t)(bt * Hh + h) * Hd + lr) * Nn + kt0 + c * 32 + 8 * g;
      v16bf vb = mk16(*(const v8bf*)vrow, *(const v8bf*)(vrow + 16));
      acc = __builtin_amdgcn_wmma_f32_16x16x32_bf16(false, pa, false, vb,
                                                    (short)0, acc, false, false);
    }
    asm volatile("" ::: "memory");  // keep next-iter LDS stores after these loads
  }
#pragma unroll
  for (int j = 0; j < 8; ++j) {
    const float v = acc[j] / l[j];
    const int row = bt * Nn + m0 + j + 8 * g;
    out[(size_t)row * Dd + h * Hd + lr] = (__bf16)v;
  }
}

// ---------------- layernorm: one wave per 128-wide row ----------------
__global__ void ln_kernel(const float* __restrict__ x, const float* __restrict__ gm,
                          const float* __restrict__ be, float* __restrict__ outf,
                          __bf16* __restrict__ outbf) {
  const int lane = threadIdx.x & 31;
  const int row = blockIdx.x * 8 + (threadIdx.x >> 5);
  const float4 xv = *(const float4*)(x + (size_t)row * Dd + lane * 4);
  float s = xv.x + xv.y + xv.z + xv.w;
#pragma unroll
  for (int msk = 1; msk < 32; msk <<= 1) s += __shfl_xor(s, msk, 32);
  const float mean = s * (1.0f / Dd);
  const float d0 = xv.x - mean, d1 = xv.y - mean, d2 = xv.z - mean, d3 = xv.w - mean;
  float sq = d0 * d0 + d1 * d1 + d2 * d2 + d3 * d3;
#pragma unroll
  for (int msk = 1; msk < 32; msk <<= 1) sq += __shfl_xor(sq, msk, 32);
  const float rstd = rsqrtf(sq * (1.0f / Dd) + 1e-5f);
  const float4 gv = *(const float4*)(gm + lane * 4);
  const float4 bvv = *(const float4*)(be + lane * 4);
  float4 y;
  y.x = d0 * rstd * gv.x + bvv.x;
  y.y = d1 * rstd * gv.y + bvv.y;
  y.z = d2 * rstd * gv.z + bvv.z;
  y.w = d3 * rstd * gv.w + bvv.w;
  *(float4*)(outf + (size_t)row * Dd + lane * 4) = y;
  if (outbf) {
    __bf16* ob = outbf + (size_t)row * Dd + lane * 4;
    ob[0] = (__bf16)y.x; ob[1] = (__bf16)y.y; ob[2] = (__bf16)y.z; ob[3] = (__bf16)y.w;
  }
}

// ---------------- launch ----------------
extern "C" void kernel_launch(void* const* d_in, const int* in_sizes, int n_in,
                              void* d_out, int out_size, void* d_ws, size_t ws_size,
                              hipStream_t stream) {
  const float* x    = (const float*)d_in[0];
  const float* Wq   = (const float*)d_in[1];
  const float* bq   = (const float*)d_in[2];
  const float* Wk   = (const float*)d_in[3];
  const float* bk   = (const float*)d_in[4];
  const float* Wv   = (const float*)d_in[5];
  const float* bvp  = (const float*)d_in[6];
  const float* Wo   = (const float*)d_in[7];
  const float* bo   = (const float*)d_in[8];
  const float* W1   = (const float*)d_in[9];
  const float* b1   = (const float*)d_in[10];
  const float* W2   = (const float*)d_in[11];
  const float* b2   = (const float*)d_in[12];
  const float* ln1g = (const float*)d_in[13];
  const float* ln1b = (const float*)d_in[14];
  const float* ln2g = (const float*)d_in[15];
  const float* ln2b = (const float*)d_in[16];
  float* out = (float*)d_out;
  (void)in_sizes; (void)n_in; (void)out_size; (void)ws_size;

  char* wsb = (char*)d_ws;
  const size_t S_xD = (size_t)Mm * Dd * 2;
  size_t off = 0;
  __bf16* xbf  = (__bf16*)(wsb + off); off += S_xD;
  __bf16* WqT  = (__bf16*)(wsb + off); off += (size_t)Dd * Dd * 2;
  __bf16* WkT  = (__bf16*)(wsb + off); off += (size_t)Dd * Dd * 2;
  __bf16* WvT  = (__bf16*)(wsb + off); off += (size_t)Dd * Dd * 2;
  __bf16* WoT  = (__bf16*)(wsb + off); off += (size_t)Dd * Dd * 2;
  __bf16* W1T  = (__bf16*)(wsb + off); off += (size_t)Ffm * Dd * 2;
  __bf16* W2T  = (__bf16*)(wsb + off); off += (size_t)Dd * Ffm * 2;
  __bf16* qbf  = (__bf16*)(wsb + off); off += S_xD;
  __bf16* kbf  = (__bf16*)(wsb + off); off += S_xD;
  __bf16* vTbf = (__bf16*)(wsb + off); off += S_xD;
  __bf16* attn = (__bf16*)(wsb + off); off += S_xD;
  float*  r1   = (float*)(wsb + off);  off += (size_t)Mm * Dd * 4;
  __bf16* y1bf = (__bf16*)(wsb + off); off += S_xD;
  __bf16* ffh  = (__bf16*)(wsb + off); off += (size_t)Mm * Ffm * 2;

  k_f32_to_bf16<<<(Mm * Dd + 255) / 256, 256, 0, stream>>>(x, xbf, Mm * Dd);
  k_transpose_bf16<<<(Dd * Dd + 255) / 256, 256, 0, stream>>>(Wq, WqT, Dd, Dd);
  k_transpose_bf16<<<(Dd * Dd + 255) / 256, 256, 0, stream>>>(Wk, WkT, Dd, Dd);
  k_transpose_bf16<<<(Dd * Dd + 255) / 256, 256, 0, stream>>>(Wv, WvT, Dd, Dd);
  k_transpose_bf16<<<(Dd * Dd + 255) / 256, 256, 0, stream>>>(Wo, WoT, Dd, Dd);
  k_transpose_bf16<<<(Dd * Ffm + 255) / 256, 256, 0, stream>>>(W1, W1T, Dd, Ffm);
  k_transpose_bf16<<<(Ffm * Dd + 255) / 256, 256, 0, stream>>>(W2, W2T, Ffm, Dd);

  const int gemmD_blocks = (Mm / 32) * (Dd / 64) / 8;   // 384
  const int gemmF_blocks = (Mm / 32) * (Ffm / 64) / 8;  // 768
  gemm_kernel<EPI_BF><<<gemmD_blocks, 256, 0, stream>>>(xbf, WqT, bq, Dd, Dd, 0.25f,
                                                        qbf, nullptr, nullptr);
  gemm_kernel<EPI_BF><<<gemmD_blocks, 256, 0, stream>>>(xbf, WkT, bk, Dd, Dd, 1.0f,
                                                        kbf, nullptr, nullptr);
  gemm_kernel<EPI_VT><<<gemmD_blocks, 256, 0, stream>>>(xbf, WvT, bvp, Dd, Dd, 1.0f,
                                                        vTbf, nullptr, nullptr);

  const int attn_blocks = (BTn * Hh * (Nn / 16)) / 8;   // 3072
  attn_kernel<<<attn_blocks, 256, 0, stream>>>(qbf, kbf, vTbf, attn);

  gemm_kernel<EPI_F32RES><<<gemmD_blocks, 256, 0, stream>>>(attn, WoT, bo, Dd, Dd, 1.0f,
                                                            nullptr, x, r1);
  ln_kernel<<<Mm / 8, 256, 0, stream>>>(r1, ln1g, ln1b, r1, y1bf);

  gemm_kernel<EPI_RELU><<<gemmF_blocks, 256, 0, stream>>>(y1bf, W1T, b1, Dd, Ffm, 1.0f,
                                                          ffh, nullptr, nullptr);
  gemm_kernel<EPI_F32RES><<<gemmD_blocks, 256, 0, stream>>>(ffh, W2T, b2, Ffm, Dd, 1.0f,
                                                            nullptr, r1, r1);

  ln_kernel<<<Mm / 8, 256, 0, stream>>>(r1, ln2g, ln2b, out, nullptr);
}
